// SLinOSSBlock_26113401160191
// MI455X (gfx1250) — matrix operand order repaired and checked by reading, not compiled, not run on hardware
//
#include <hip/hip_runtime.h>
#include <hip/hip_bf16.h>
#include <string.h>

// ---------------- model constants ----------------
#define BATCH     4
#define SEQ       2048
#define DMODEL    1024
#define DSTATE    16
#define DHEAD     64
#define DINNER    2048            // EXPAND * DMODEL
#define NHEADS    32              // DINNER / DHEAD
#define DCONV     4
#define FFNDIM    2816
#define PROJDIM   4160            // 2*DINNER + 2*DSTATE + NHEADS
#define CONVCH    2080            // DINNER + 2*DSTATE
#define MROWS     (BATCH*SEQ)     // 8192
#define EPSF      1e-6f

// ---------------- bf16 helpers (raw u16 storage) ----------------
__device__ __forceinline__ unsigned short f2bf(float f) {
  unsigned int u = __float_as_uint(f);
  unsigned int r = (u + 0x7FFFu + ((u >> 16) & 1u)) >> 16;   // round-nearest-even
  return (unsigned short)r;
}
__device__ __forceinline__ float bf2f(unsigned short h) {
  return __uint_as_float(((unsigned int)h) << 16);
}

typedef __attribute__((ext_vector_type(16))) __bf16 v16bf;
typedef __attribute__((ext_vector_type(8)))  float  v8f;

union FragBF { uint4 u[2]; v16bf v; };

// ---------------- generic f32 -> bf16 convert ----------------
__global__ void k_f32_to_bf16(const float* __restrict__ in,
                              unsigned short* __restrict__ out, size_t n) {
  size_t i = (size_t)blockIdx.x * blockDim.x + threadIdx.x;
  if (i < n) out[i] = f2bf(in[i]);
}

// ---------------- RMSNorm (row of C) -> bf16 ----------------
__global__ __launch_bounds__(256)
void k_rmsnorm_bf16(const float* __restrict__ x, const float* __restrict__ w,
                    unsigned short* __restrict__ out, int C) {
  int row = blockIdx.x;
  size_t base = (size_t)row * C;
  float s = 0.f;
  for (int c = threadIdx.x; c < C; c += 256) { float v = x[base + c]; s += v * v; }
  __shared__ float red[8];
  for (int o = 16; o > 0; o >>= 1) s += __shfl_xor(s, o);
  if ((threadIdx.x & 31) == 0) red[threadIdx.x >> 5] = s;
  __syncthreads();
  if (threadIdx.x < 8) {
    float t = red[threadIdx.x];
    for (int o = 4; o > 0; o >>= 1) t += __shfl_xor(t, o);
    if (threadIdx.x == 0) red[0] = t;
  }
  __syncthreads();
  float scale = rsqrtf(red[0] / (float)C + EPSF);
  for (int c = threadIdx.x; c < C; c += 256)
    out[base + c] = f2bf(x[base + c] * scale * w[c]);
}

// ---------------- residual add + RMSNorm -> h (f32) and hn (bf16) --------
__global__ __launch_bounds__(256)
void k_resid_rmsnorm(const float* __restrict__ x, const float* __restrict__ mix,
                     const float* __restrict__ w, float* __restrict__ h,
                     unsigned short* __restrict__ hn, int C) {
  int row = blockIdx.x;
  size_t base = (size_t)row * C;
  float s = 0.f;
  for (int c = threadIdx.x; c < C; c += 256) {
    float v = x[base + c] + mix[base + c];
    h[base + c] = v;
    s += v * v;
  }
  __shared__ float red[8];
  for (int o = 16; o > 0; o >>= 1) s += __shfl_xor(s, o);
  if ((threadIdx.x & 31) == 0) red[threadIdx.x >> 5] = s;
  __syncthreads();
  if (threadIdx.x < 8) {
    float t = red[threadIdx.x];
    for (int o = 4; o > 0; o >>= 1) t += __shfl_xor(t, o);
    if (threadIdx.x == 0) red[0] = t;
  }
  __syncthreads();
  float scale = rsqrtf(red[0] / (float)C + EPSF);
  for (int c = threadIdx.x; c < C; c += 256)
    hn[base + c] = f2bf(h[base + c] * scale * w[c]);
}

// ---------------- WMMA bf16 GEMM:  out[M,N] = A[M,K] * W[N,K]^T ----------
// Register-blocked: each wave owns a 64x64 output tile = 4 (M) x 4 (N)
// 16x16 WMMA tiles. Per 32-wide k-chunk: 16 b128 loads feed 16
// v_wmma_f32_16x16x32_bf16 ops (each A and B fragment reused 4x,
// ~32 FLOP per byte of cache traffic).
//
// A fragment (16x32 bf16): lane = 16*half + m; elems 0-7  -> K = kc+8*half+0..7
//                                              elems 8-15 -> K = kc+16+8*half+0..7
// B fragment (32x16 bf16): lane = 16*half + n; elems 0-7  -> K = kc+8*half+0..7
//                                              elems 8-15 -> K = kc+16+8*half+0..7
// D (f32 16x16): vgpr r, lane = 16*half + n  -> row = r + 8*half, col = n
template <int OUT_BF16>
__global__ __launch_bounds__(256)
void k_gemm_wmma_bf16(const unsigned short* __restrict__ A,
                      const unsigned short* __restrict__ W,
                      float* __restrict__ outF,
                      unsigned short* __restrict__ outB,
                      int M, int N, int K) {
  const int wave = threadIdx.x >> 5;          // 0..7
  const int lane = threadIdx.x & 31;
  const int tilesN = N >> 6;                  // 64-wide N tiles
  const int tilesM = M >> 6;                  // 64-tall M tiles
  const int tile = blockIdx.x * 8 + wave;
  if (tile >= tilesM * tilesN) return;        // whole-wave guard: EXEC all-ones
  const int tm = tile / tilesN;
  const int tn = tile - tm * tilesN;
  const int half = lane >> 4;
  const int lan15 = lane & 15;
  const int m0 = tm << 6;
  const int n0 = tn << 6;

  const unsigned short* aP[4];
  const unsigned short* bP[4];
#pragma unroll
  for (int i = 0; i < 4; ++i) {
    aP[i] = A + (size_t)(m0 + i * 16 + lan15) * K + half * 8;
    bP[i] = W + (size_t)(n0 + i * 16 + lan15) * K + half * 8;
  }

  v8f acc[4][4];
#pragma unroll
  for (int mi = 0; mi < 4; ++mi)
#pragma unroll
    for (int ni = 0; ni < 4; ++ni)
#pragma unroll
      for (int i = 0; i < 8; ++i) acc[mi][ni][i] = 0.f;

  for (int kc = 0; kc < K; kc += 32) {
    FragBF fa[4], fb[4];
#pragma unroll
    for (int i = 0; i < 4; ++i) {
      fa[i].u[0] = *(const uint4*)(const void*)(aP[i] + kc);
      fa[i].u[1] = *(const uint4*)(const void*)(aP[i] + kc + 16);
    }
#pragma unroll
    for (int i = 0; i < 4; ++i) {
      fb[i].u[0] = *(const uint4*)(const void*)(bP[i] + kc);
      fb[i].u[1] = *(const uint4*)(const void*)(bP[i] + kc + 16);
    }
#pragma unroll
    for (int mi = 0; mi < 4; ++mi)
#pragma unroll
      for (int ni = 0; ni < 4; ++ni)
        acc[mi][ni] = __builtin_amdgcn_wmma_f32_16x16x32_bf16(
            false, fa[mi].v, false, fb[ni].v, (short)0, acc[mi][ni], false, false);
  }

#pragma unroll
  for (int mi = 0; mi < 4; ++mi) {
#pragma unroll
    for (int ni = 0; ni < 4; ++ni) {
      const size_t base = (size_t)(m0 + mi * 16) * N + (n0 + ni * 16) + lan15;
#pragma unroll
      for (int r = 0; r < 8; ++r) {
        int row = r + half * 8;
        if (OUT_BF16) outB[base + (size_t)row * N] = f2bf(acc[mi][ni][r]);
        else          outF[base + (size_t)row * N] = acc[mi][ni][r];
      }
    }
  }
}

// ---------------- depthwise causal conv(4) + SiLU ----------------
// reads xBC columns [2048,4128) of zxbcdt; writes xbc_act (MROWS x CONVCH)
__global__ void k_conv_silu(const float* __restrict__ zx,
                            const float* __restrict__ cw,
                            const float* __restrict__ cb,
                            float* __restrict__ out) {
  size_t idx = (size_t)blockIdx.x * blockDim.x + threadIdx.x;
  const size_t total = (size_t)MROWS * CONVCH;
  if (idx >= total) return;
  int c = (int)(idx % CONVCH);
  int row = (int)(idx / CONVCH);
  int l = row & (SEQ - 1);
  float acc = cb[c];
#pragma unroll
  for (int k = 0; k < DCONV; ++k) {
    int ls = l + k - (DCONV - 1);
    if (ls >= 0)
      acc += zx[(size_t)(row + k - (DCONV - 1)) * PROJDIM + DINNER + c] * cw[c * DCONV + k];
  }
  out[idx] = acc / (1.f + expf(-acc));   // SiLU
}

// ---------------- SSM scan ----------------
// grid = BATCH*NHEADS blocks, 64 threads (one per d). 16 states in registers.
// Fuses D*u skip and silu(z) gate; writes bf16 gated output for out_proj GEMM.
__global__ __launch_bounds__(64)
void k_ssm_scan(const float* __restrict__ zx,     // MROWS x PROJDIM (z, -, dt_raw)
                const float* __restrict__ xbc,    // MROWS x CONVCH  (u, B, C)
                const float* __restrict__ A_log,  // NHEADS x DSTATE
                const float* __restrict__ dt_bias,// NHEADS
                const float* __restrict__ Dp,     // NHEADS
                unsigned short* __restrict__ ygate) { // MROWS x DINNER (bf16)
  const int bh = blockIdx.x;
  const int b = bh >> 5;           // / NHEADS
  const int h = bh & 31;
  const int d = threadIdx.x;
  __shared__ float sB[DSTATE], sC[DSTATE], sdt;

  float A[DSTATE], w[DSTATE], ys[DSTATE];
#pragma unroll
  for (int n = 0; n < DSTATE; ++n) {
    A[n] = expf(A_log[h * DSTATE + n]);
    w[n] = 0.f; ys[n] = 0.f;
  }
  const float Dh = Dp[h];
  const float dtb = dt_bias[h];

  for (int t = 0; t < SEQ; ++t) {
    const size_t row = (size_t)b * SEQ + t;
    if (d < DSTATE)            sB[d] = xbc[row * CONVCH + DINNER + d];
    else if (d < 2 * DSTATE)   sC[d - DSTATE] = xbc[row * CONVCH + DINNER + DSTATE + (d - DSTATE)];
    else if (d == 2 * DSTATE) {
      float dr = zx[row * PROJDIM + 2 * DINNER + 2 * DSTATE + h] + dtb;
      sdt = (dr > 20.f) ? dr : log1pf(expf(dr));   // softplus
    }
    __syncthreads();

    float nb = 0.f, nc = 0.f;
#pragma unroll
    for (int n = 0; n < DSTATE; ++n) { nb += sB[n] * sB[n]; nc += sC[n] * sC[n]; }
    const float ib = 1.f / (sqrtf(nb) + EPSF);
    const float ic = 1.f / (sqrtf(nc) + EPSF);
    const float dt = sdt;
    const float dt2 = dt * dt;
    const float u = xbc[row * CONVCH + h * DHEAD + d];

    float o = 0.f;
#pragma unroll
    for (int n = 0; n < DSTATE; ++n) {
      const float An = A[n];
      const float S  = 1.f / (1.f + dt2 * An);
      const float Bu = sB[n] * ib * u;
      w[n] = S * (w[n] - dt * An * ys[n] + dt * Bu);
      ys[n] += dt * w[n];
      o += ys[n] * (sC[n] * ic);
    }
    const float z = zx[row * PROJDIM + h * DHEAD + d];
    const float g = z / (1.f + expf(-z));          // silu(z)
    ygate[row * DINNER + h * DHEAD + d] = f2bf((o + Dh * u) * g);
    __syncthreads();
  }
}

// ---------------- silu(g) * u -> bf16 ----------------
__global__ void k_silu_mul(const unsigned short* __restrict__ g,
                           const unsigned short* __restrict__ u,
                           unsigned short* __restrict__ t, size_t n) {
  size_t i = (size_t)blockIdx.x * blockDim.x + threadIdx.x;
  if (i >= n) return;
  float gv = bf2f(g[i]);
  float uv = bf2f(u[i]);
  t[i] = f2bf(gv / (1.f + expf(-gv)) * uv);
}

// ---------------- final residual add (f32) ----------------
__global__ void k_add(const float* __restrict__ a, const float* __restrict__ b,
                      float* __restrict__ out, size_t n) {
  size_t i = (size_t)blockIdx.x * blockDim.x + threadIdx.x;
  if (i < n) out[i] = a[i] + b[i];
}

// ---------------- host orchestration ----------------
static inline size_t alignUp(size_t x) { return (x + 255) & ~(size_t)255; }

extern "C" void kernel_launch(void* const* d_in, const int* in_sizes, int n_in,
                              void* d_out, int out_size, void* d_ws, size_t ws_size,
                              hipStream_t stream) {
  (void)in_sizes; (void)n_in; (void)out_size; (void)ws_size;

  const float* x        = (const float*)d_in[0];   // (4,2048,1024)
  const float* norm1_w  = (const float*)d_in[1];
  const float* in_proj  = (const float*)d_in[2];   // (4160,1024)
  const float* conv_w   = (const float*)d_in[3];   // (2080,4)
  const float* conv_b   = (const float*)d_in[4];
  const float* A_log    = (const float*)d_in[5];   // (32,16)
  const float* dt_bias  = (const float*)d_in[6];
  const float* Dp       = (const float*)d_in[7];
  const float* out_proj = (const float*)d_in[8];   // (1024,2048)
  const float* norm2_w  = (const float*)d_in[9];
  const float* gate_w   = (const float*)d_in[10];  // (2816,1024)
  const float* up_w     = (const float*)d_in[11];  // (2816,1024)
  const float* down_w   = (const float*)d_in[12];  // (1024,2816)
  float* out            = (float*)d_out;

  // workspace carve-up
  char* ws = (char*)d_ws;
  size_t off = 0;
  auto carve = [&](size_t bytes) { char* p = ws + off; off = alignUp(off + bytes); return p; };

  unsigned short* xn_bf     = (unsigned short*)carve((size_t)MROWS * DMODEL * 2);
  unsigned short* inproj_bf = (unsigned short*)carve((size_t)PROJDIM * DMODEL * 2);
  unsigned short* outproj_bf= (unsigned short*)carve((size_t)DMODEL * DINNER * 2);
  unsigned short* gate_bf   = (unsigned short*)carve((size_t)FFNDIM * DMODEL * 2);
  unsigned short* up_bf     = (unsigned short*)carve((size_t)FFNDIM * DMODEL * 2);
  unsigned short* down_bf   = (unsigned short*)carve((size_t)DMODEL * FFNDIM * 2);
  float*          zxbcdt    = (float*)carve((size_t)MROWS * PROJDIM * 4);
  float*          xbc_act   = (float*)carve((size_t)MROWS * CONVCH * 4);
  unsigned short* ygate_bf  = (unsigned short*)carve((size_t)MROWS * DINNER * 2);
  float*          mix       = (float*)carve((size_t)MROWS * DMODEL * 4);
  float*          hbuf      = (float*)carve((size_t)MROWS * DMODEL * 4);
  unsigned short* hn_bf     = (unsigned short*)carve((size_t)MROWS * DMODEL * 2);
  unsigned short* g_bf      = (unsigned short*)carve((size_t)MROWS * FFNDIM * 2);
  unsigned short* u_bf      = (unsigned short*)carve((size_t)MROWS * FFNDIM * 2);
  unsigned short* t_bf      = (unsigned short*)carve((size_t)MROWS * FFNDIM * 2);
  float*          ffn       = (float*)carve((size_t)MROWS * DMODEL * 4);

  auto cvt = [&](const float* src, unsigned short* dst, size_t n) {
    k_f32_to_bf16<<<(unsigned)((n + 255) / 256), 256, 0, stream>>>(src, dst, n);
  };

  // 0) weights -> bf16
  cvt(in_proj,  inproj_bf,  (size_t)PROJDIM * DMODEL);
  cvt(out_proj, outproj_bf, (size_t)DMODEL * DINNER);
  cvt(gate_w,   gate_bf,    (size_t)FFNDIM * DMODEL);
  cvt(up_w,     up_bf,      (size_t)FFNDIM * DMODEL);
  cvt(down_w,   down_bf,    (size_t)DMODEL * FFNDIM);

  // 1) RMSNorm1 -> bf16
  k_rmsnorm_bf16<<<MROWS, 256, 0, stream>>>(x, norm1_w, xn_bf, DMODEL);

  // 2) in_proj GEMM: (8192x1024) x (4160x1024)^T -> zxbcdt f32
  {
    int tiles = (MROWS >> 6) * (PROJDIM >> 6);
    k_gemm_wmma_bf16<0><<<(tiles + 7) / 8, 256, 0, stream>>>(
        xn_bf, inproj_bf, zxbcdt, nullptr, MROWS, PROJDIM, DMODEL);
  }

  // 3) conv + silu
  {
    size_t n = (size_t)MROWS * CONVCH;
    k_conv_silu<<<(unsigned)((n + 255) / 256), 256, 0, stream>>>(zxbcdt, conv_w, conv_b, xbc_act);
  }

  // 4) SSM scan (fused D-skip and silu(z) gate) -> ygate bf16
  k_ssm_scan<<<BATCH * NHEADS, 64, 0, stream>>>(zxbcdt, xbc_act, A_log, dt_bias, Dp, ygate_bf);

  // 5) out_proj GEMM: (8192x2048) x (1024x2048)^T -> mix f32
  {
    int tiles = (MROWS >> 6) * (DMODEL >> 6);
    k_gemm_wmma_bf16<0><<<(tiles + 7) / 8, 256, 0, stream>>>(
        ygate_bf, outproj_bf, mix, nullptr, MROWS, DMODEL, DINNER);
  }

  // 6) h = x + mix ; hn = rmsnorm(h)*norm2_w -> bf16
  k_resid_rmsnorm<<<MROWS, 256, 0, stream>>>(x, mix, norm2_w, hbuf, hn_bf, DMODEL);

  // 7) FFN gate & up GEMMs -> bf16
  {
    int tiles = (MROWS >> 6) * (FFNDIM >> 6);
    k_gemm_wmma_bf16<1><<<(tiles + 7) / 8, 256, 0, stream>>>(
        hn_bf, gate_bf, nullptr, g_bf, MROWS, FFNDIM, DMODEL);
    k_gemm_wmma_bf16<1><<<(tiles + 7) / 8, 256, 0, stream>>>(
        hn_bf, up_bf, nullptr, u_bf, MROWS, FFNDIM, DMODEL);
  }

  // 8) t = silu(g) * u  -> bf16
  {
    size_t n = (size_t)MROWS * FFNDIM;
    k_silu_mul<<<(unsigned)((n + 255) / 256), 256, 0, stream>>>(g_bf, u_bf, t_bf, n);
  }

  // 9) down GEMM: (8192x2816) x (1024x2816)^T -> ffn f32
  {
    int tiles = (MROWS >> 6) * (DMODEL >> 6);
    k_gemm_wmma_bf16<0><<<(tiles + 7) / 8, 256, 0, stream>>>(
        t_bf, down_bf, ffn, nullptr, MROWS, DMODEL, FFNDIM);
  }

  // 10) out = h + ffn
  {
    size_t n = (size_t)MROWS * DMODEL;
    k_add<<<(unsigned)((n + 255) / 256), 256, 0, stream>>>(hbuf, ffn, out, n);
  }
}
